// MultiHeadAttention_46394236731623
// MI455X (gfx1250) — matrix operand order repaired and checked by reading, not compiled, not run on hardware
//
#include <hip/hip_runtime.h>

// ---------------------------------------------------------------------------
// MultiHeadAttention for gfx1250 (MI455X): bf16 WMMA everywhere.
//   B=4, L=1024, D_MODEL=1024, H=16, D_K=64, M = B*L = 4096.
// Pipeline:
//   1) cvt_f32_to_bf16      : query/key/value -> bf16 [M,1024]
//   2) transpose_to_bf16    : w{q,k,v,o} [K,N] f32 -> [N,K] bf16 (WMMA B wants K-contiguous)
//   3) gemm_bf16_wmma mode0 : Q,K bf16 [M,1024] = X @ W + b
//      gemm_bf16_wmma mode1 : V bf16 stored transposed [B,H,64,L]
//   4) attn_wmma            : flash-style attention, out bf16 [M,1024]
//   5) gemm_bf16_wmma mode2 : f32 output = Att @ Wo + bo
// GEMM: 64x32 tile per wave (4x2 WMMA accums) -> ~21 FLOP/byte from cache.
// ---------------------------------------------------------------------------

typedef __bf16 bf16_t;
typedef __attribute__((ext_vector_type(16))) __bf16 v16bf;
typedef __attribute__((ext_vector_type(8)))  __bf16 v8bf;
typedef __attribute__((ext_vector_type(4)))  __bf16 v4bf;
typedef __attribute__((ext_vector_type(8)))  float  v8f;

#define LSEQ 1024
#define DMODEL 1024
#define NHEAD 16
#define DK 64
#define MROWS 4096

__device__ __forceinline__ v8f wmma_bf16(v16bf a, v16bf b, v8f c) {
  // (neg_a, A, neg_b, B, c_mod, C, reuse_a, reuse_b)
  return __builtin_amdgcn_wmma_f32_16x16x32_bf16(false, a, false, b, (short)0, c,
                                                 false, false);
}

// A-matrix fragment (16x32 bf16). Lane l (<16): K = {0..15};
// lanes >=16 use koff=8 -> K = {16..31}. `p` points at row start + k-chunk.
__device__ __forceinline__ v16bf load_a_frag(const bf16_t* p, int koff) {
  v8bf lo = *reinterpret_cast<const v8bf*>(p + 2 * koff);
  v8bf hi = *reinterpret_cast<const v8bf*>(p + 2 * koff + 8);
  v16bf a;
#pragma unroll
  for (int i = 0; i < 8; ++i) { a[i] = lo[i]; a[i + 8] = hi[i]; }
  return a;
}

__device__ __forceinline__ v8f zero_v8f() {
  v8f z = {0.f, 0.f, 0.f, 0.f, 0.f, 0.f, 0.f, 0.f};
  return z;
}

// ---------------------------------------------------------------------------
// fp32 -> bf16 elementwise (vectorized x4)
// ---------------------------------------------------------------------------
__global__ __launch_bounds__(256) void cvt_f32_to_bf16(const float* __restrict__ src,
                                                       bf16_t* __restrict__ dst,
                                                       int n4) {
  int i = blockIdx.x * 256 + threadIdx.x;
  if (i < n4) {
    float4 v = reinterpret_cast<const float4*>(src)[i];
    v4bf o;
    o[0] = (bf16_t)v.x; o[1] = (bf16_t)v.y; o[2] = (bf16_t)v.z; o[3] = (bf16_t)v.w;
    reinterpret_cast<v4bf*>(dst)[i] = o;
  }
}

// ---------------------------------------------------------------------------
// [1024,1024] f32 -> transposed [1024,1024] bf16 (LDS-tiled, coalesced)
// ---------------------------------------------------------------------------
__global__ __launch_bounds__(256) void transpose_to_bf16(const float* __restrict__ src,
                                                         bf16_t* __restrict__ dst) {
  __shared__ float tile[32][33];
  int tx = threadIdx.x & 31;
  int ty = threadIdx.x >> 5;           // 0..7
  int bx = blockIdx.x & 31;            // col tile of src
  int by = blockIdx.x >> 5;            // row tile of src
  int col = bx * 32 + tx;
#pragma unroll
  for (int i = 0; i < 32; i += 8) {
    int row = by * 32 + ty + i;
    tile[ty + i][tx] = src[(size_t)row * DMODEL + col];
  }
  __syncthreads();
  int ocol = by * 32 + tx;             // dst col = src row
#pragma unroll
  for (int i = 0; i < 32; i += 8) {
    int orow = bx * 32 + ty + i;       // dst row = src col
    dst[(size_t)orow * DMODEL + ocol] = (bf16_t)tile[tx][ty + i];
  }
}

// ---------------------------------------------------------------------------
// GEMM: out[M=4096, N=1024] = A[M,1024] @ Bt^T + bias, bf16 WMMA, f32 accum.
//   A  : bf16 row-major [4096,1024]
//   Bt : bf16 [1024,1024] = W^T (row n = output column n, K-contiguous)
// mode 0: out bf16 [M,1024]
// mode 1: out bf16 V-transposed [B,H,64,L]
// mode 2: out f32  [M,1024]
// One wave computes a 64x32 tile (4x2 WMMA accumulators). 4 waves / block
// cover 64x128 (same A rows -> shared via L0/L1).
// ---------------------------------------------------------------------------
__global__ __launch_bounds__(128) void gemm_bf16_wmma(const bf16_t* __restrict__ A,
                                                      const bf16_t* __restrict__ Bt,
                                                      const float* __restrict__ bias,
                                                      void* __restrict__ out,
                                                      int mode) {
  const int wave = threadIdx.x >> 5;
  const int lane = threadIdx.x & 31;
  const int lh = lane & 15;
  const bool hiHalf = lane >= 16;

  const int tile = blockIdx.x * 4 + wave;        // 2048 tiles total
  const int nt = tile & 31;                      // N/32  = 32
  const int mt = tile >> 5;                      // M/64  = 64
  const int m0 = mt * 64, n0 = nt * 32;

  v8f c[4][2];
#pragma unroll
  for (int mi = 0; mi < 4; ++mi) { c[mi][0] = zero_v8f(); c[mi][1] = zero_v8f(); }

  const bf16_t* arow[4];
#pragma unroll
  for (int mi = 0; mi < 4; ++mi)
    arow[mi] = A + (size_t)(m0 + mi * 16 + lh) * DMODEL;
  const bf16_t* brow0 = Bt + (size_t)(n0 + lh) * DMODEL;
  const bf16_t* brow1 = Bt + (size_t)(n0 + 16 + lh) * DMODEL;
  const int aoff = hiHalf ? 8 : 0;
  const int boff = hiHalf ? 16 : 0;

  for (int kk = 0; kk < DMODEL; kk += 32) {
    v16bf b0 = *reinterpret_cast<const v16bf*>(brow0 + kk + boff);
    v16bf b1 = *reinterpret_cast<const v16bf*>(brow1 + kk + boff);
#pragma unroll
    for (int mi = 0; mi < 4; ++mi) {
      v16bf a = load_a_frag(arow[mi] + kk, aoff);
      c[mi][0] = wmma_bf16(a, b0, c[mi][0]);
      c[mi][1] = wmma_bf16(a, b1, c[mi][1]);
    }
  }

#pragma unroll
  for (int mi = 0; mi < 4; ++mi) {
#pragma unroll
    for (int nj = 0; nj < 2; ++nj) {
      const int col = n0 + nj * 16 + lh;
      const float bv = bias[col];
#pragma unroll
      for (int r = 0; r < 8; ++r) {
        const int row = m0 + mi * 16 + r + (hiHalf ? 8 : 0);
        const float val = c[mi][nj][r] + bv;
        if (mode == 0) {
          ((bf16_t*)out)[(size_t)row * DMODEL + col] = (bf16_t)val;
        } else if (mode == 1) {
          const int bb = row >> 10, l = row & 1023;   // L = 1024
          const int hh = col >> 6, dd = col & 63;     // DK = 64
          ((bf16_t*)out)[((size_t)((bb * NHEAD + hh) * DK + dd)) * LSEQ + l] =
              (bf16_t)val;
        } else {
          ((float*)out)[(size_t)row * DMODEL + col] = val;
        }
      }
    }
  }
}

// ---------------------------------------------------------------------------
// Flash attention. One wave handles 16 query rows x one head, full D_K=64.
//   Qb,Kb : bf16 [B,L,H*DK]   Vt : bf16 [B,H,DK,L]   mask : int32 [B,1,L,L]
//   Att   : bf16 [B,L,H*DK]
// Per 32-key chunk: 4 score WMMAs, online softmax in registers (C-layout
// vgpr r <-> row r / r+8 matches per-row stats exactly), P staged through
// LDS to re-layout into an A fragment, then 4 P@V WMMAs.
// ---------------------------------------------------------------------------
__global__ __launch_bounds__(128) void attn_wmma(const bf16_t* __restrict__ Qb,
                                                 const bf16_t* __restrict__ Kb,
                                                 const bf16_t* __restrict__ Vt,
                                                 const int* __restrict__ mask,
                                                 bf16_t* __restrict__ Att) {
  __shared__ bf16_t plds[4][16 * 40];   // per-wave P tile, row stride 40 (pad)
  const int wave = threadIdx.x >> 5;
  const int lane = threadIdx.x & 31;
  const int lh = lane & 15;
  const bool hiHalf = lane >= 16;
  const int hi8 = hiHalf ? 8 : 0;

  const int id = blockIdx.x * 4 + wave;  // 4096 waves total
  const int qt = id & 63;                // L/16 = 64 q-tiles
  const int h = (id >> 6) & 15;
  const int b = id >> 10;
  const int q0 = qt * 16;

  const int aoff = hiHalf ? 8 : 0;
  // Q fragments resident for the whole key loop: 16x64 = two 16x32 A frags.
  const bf16_t* qrow = Qb + (size_t)(b * LSEQ + q0 + lh) * DMODEL + h * DK;
  v16bf qa0 = load_a_frag(qrow, aoff);
  v16bf qa1 = load_a_frag(qrow + 32, aoff);

  v8f o[4];
  float mrow[8], lrow[8];
#pragma unroll
  for (int j = 0; j < 4; ++j) o[j] = zero_v8f();
#pragma unroll
  for (int r = 0; r < 8; ++r) { mrow[r] = -__builtin_inff(); lrow[r] = 0.f; }

  bf16_t* pl = plds[wave];
  const size_t maskbase = ((size_t)b * LSEQ) * LSEQ;

  for (int k0 = 0; k0 < LSEQ; k0 += 32) {
    // ---- scores: two 16x16 tiles over keys [k0,k0+16) and [k0+16,k0+32)
    v8f s[2];
#pragma unroll
    for (int t = 0; t < 2; ++t) {
      const int key = k0 + t * 16 + lh;
      const bf16_t* krow =
          Kb + (size_t)(b * LSEQ + key) * DMODEL + h * DK + (hiHalf ? 16 : 0);
      v16bf kb0 = *reinterpret_cast<const v16bf*>(krow);
      v16bf kb1 = *reinterpret_cast<const v16bf*>(krow + 32);
      v8f acc = zero_v8f();
      acc = wmma_bf16(qa0, kb0, acc);
      acc = wmma_bf16(qa1, kb1, acc);
      s[t] = acc;
    }
    // ---- scale 1/sqrt(DK) + mask
#pragma unroll
    for (int t = 0; t < 2; ++t) {
      const int key = k0 + t * 16 + lh;
#pragma unroll
      for (int r = 0; r < 8; ++r) {
        const int q = q0 + r + hi8;
        const int mv = mask[maskbase + (size_t)q * LSEQ + key];
        const float v = s[t][r] * 0.125f;
        s[t][r] = (mv == 0) ? -1.0e9f : v;
      }
    }
    // ---- online softmax: row max over this 32-key chunk
    float mnew[8];
#pragma unroll
    for (int r = 0; r < 8; ++r) mnew[r] = fmaxf(s[0][r], s[1][r]);
#pragma unroll
    for (int d = 1; d < 16; d <<= 1)
#pragma unroll
      for (int r = 0; r < 8; ++r)
        mnew[r] = fmaxf(mnew[r], __shfl_xor(mnew[r], d, 32));

    float sf[8];
#pragma unroll
    for (int r = 0; r < 8; ++r) {
      const float mold = mrow[r];
      const float mn = fmaxf(mold, mnew[r]);
      mrow[r] = mn;
      sf[r] = __expf(mold - mn);     // exp(-inf)=0 on first chunk
      lrow[r] *= sf[r];
    }
#pragma unroll
    for (int j = 0; j < 4; ++j)
#pragma unroll
      for (int r = 0; r < 8; ++r) o[j][r] *= sf[r];

    // ---- p = exp(s-m): accumulate row sums, stash bf16 P into LDS
    float rsum[8];
#pragma unroll
    for (int r = 0; r < 8; ++r) rsum[r] = 0.f;
#pragma unroll
    for (int t = 0; t < 2; ++t)
#pragma unroll
      for (int r = 0; r < 8; ++r) {
        const float p = __expf(s[t][r] - mrow[r]);
        rsum[r] += p;
        pl[(r + hi8) * 40 + t * 16 + lh] = (bf16_t)p;
      }
#pragma unroll
    for (int d = 1; d < 16; d <<= 1)
#pragma unroll
      for (int r = 0; r < 8; ++r) rsum[r] += __shfl_xor(rsum[r], d, 32);
#pragma unroll
    for (int r = 0; r < 8; ++r) lrow[r] += rsum[r];

    // DS ops are in-order within a wave; stop the compiler reordering and
    // make the cross-lane LDS data visible before the A-fragment reads.
    asm volatile("s_wait_dscnt 0" ::: "memory");
    v16bf pa = load_a_frag(pl + lh * 40, aoff);
    asm volatile("" ::: "memory");

    // ---- O += P @ V   (V transposed: key index contiguous)
#pragma unroll
    for (int j = 0; j < 4; ++j) {
      const bf16_t* vrow = Vt +
          ((size_t)((b * NHEAD + h) * DK + j * 16 + lh)) * LSEQ + k0 +
          (hiHalf ? 16 : 0);
      v16bf vb = *reinterpret_cast<const v16bf*>(vrow);
      o[j] = wmma_bf16(pa, vb, o[j]);
    }
  }

  // ---- epilogue: O / l, write bf16 [B,L,H*DK]
#pragma unroll
  for (int r = 0; r < 8; ++r) {
    const int q = q0 + r + hi8;
    const float inv = 1.0f / lrow[r];
    bf16_t* orow = Att + (size_t)(b * LSEQ + q) * DMODEL + h * DK;
#pragma unroll
    for (int j = 0; j < 4; ++j)
      orow[j * 16 + lh] = (bf16_t)(o[j][r] * inv);
  }
}

// ---------------------------------------------------------------------------
extern "C" void kernel_launch(void* const* d_in, const int* in_sizes, int n_in,
                              void* d_out, int out_size, void* d_ws, size_t ws_size,
                              hipStream_t stream) {
  const float* q  = (const float*)d_in[0];
  const float* k  = (const float*)d_in[1];
  const float* v  = (const float*)d_in[2];
  const int*  msk = (const int*)d_in[3];
  const float* wq = (const float*)d_in[4];
  const float* bq = (const float*)d_in[5];
  const float* wk = (const float*)d_in[6];
  const float* bk = (const float*)d_in[7];
  const float* wv = (const float*)d_in[8];
  const float* bv = (const float*)d_in[9];
  const float* wo = (const float*)d_in[10];
  const float* bo = (const float*)d_in[11];

  char* ws = (char*)d_ws;
  const size_t MB = 1u << 20;
  bf16_t* Xq  = (bf16_t*)(ws + 0 * MB);    // 8 MiB each
  bf16_t* Xk  = (bf16_t*)(ws + 8 * MB);
  bf16_t* Xv  = (bf16_t*)(ws + 16 * MB);
  bf16_t* WqT = (bf16_t*)(ws + 24 * MB);   // 2 MiB each
  bf16_t* WkT = (bf16_t*)(ws + 26 * MB);
  bf16_t* WvT = (bf16_t*)(ws + 28 * MB);
  bf16_t* WoT = (bf16_t*)(ws + 30 * MB);
  bf16_t* Qb  = (bf16_t*)(ws + 32 * MB);
  bf16_t* Kb  = (bf16_t*)(ws + 40 * MB);
  bf16_t* Vt  = (bf16_t*)(ws + 48 * MB);
  bf16_t* Att = (bf16_t*)(ws + 56 * MB);

  const int n4 = (MROWS * DMODEL) / 4;     // 1,048,576
  cvt_f32_to_bf16<<<4096, 256, 0, stream>>>(q, Xq, n4);
  cvt_f32_to_bf16<<<4096, 256, 0, stream>>>(k, Xk, n4);
  cvt_f32_to_bf16<<<4096, 256, 0, stream>>>(v, Xv, n4);

  transpose_to_bf16<<<1024, 256, 0, stream>>>(wq, WqT);
  transpose_to_bf16<<<1024, 256, 0, stream>>>(wk, WkT);
  transpose_to_bf16<<<1024, 256, 0, stream>>>(wv, WvT);
  transpose_to_bf16<<<1024, 256, 0, stream>>>(wo, WoT);

  gemm_bf16_wmma<<<512, 128, 0, stream>>>(Xq, WqT, bq, (void*)Qb, 0);
  gemm_bf16_wmma<<<512, 128, 0, stream>>>(Xk, WkT, bk, (void*)Kb, 0);
  gemm_bf16_wmma<<<512, 128, 0, stream>>>(Xv, WvT, bv, (void*)Vt, 1);

  attn_wmma<<<1024, 128, 0, stream>>>(Qb, Kb, Vt, msk, Att);

  gemm_bf16_wmma<<<512, 128, 0, stream>>>(Att, WoT, bo, d_out, 2);
}
